// QBN_GRU_Base_58110907515068
// MI455X (gfx1250) — compile-verified, matrix operand order checked by reading, not run on hardware
//
#include <hip/hip_runtime.h>
#include <hip/hip_bf16.h>
#include <cmath>

typedef __attribute__((ext_vector_type(16))) _Float16 v16h;
typedef __attribute__((ext_vector_type(8)))  float    v8f;

#define TT 512
#define BB 2048
#define IN_DIM 64
#define OBS_MID 28
#define OBS_BN 8
#define M_SIZE 16

// ---------------- CDNA5 hardware transcendentals ----------------------------
// gfx1250 has V_TANH_F32 (TRANS32 class). Prefer the clang builtin when it
// exists; otherwise emit the instruction directly. Branch-free, saturating.
__device__ __forceinline__ float fast_tanh(float x) {
#if __has_builtin(__builtin_amdgcn_tanhf)
    return __builtin_amdgcn_tanhf(x);
#elif __has_builtin(__builtin_amdgcn_tanh_f32)
    return __builtin_amdgcn_tanh_f32(x);
#else
    float r;
    asm volatile("v_tanh_f32 %0, %1" : "=v"(r) : "v"(x));
    return r;
#endif
}
__device__ __forceinline__ float sigm(float x) {
    // sigmoid(x) = 0.5*tanh(x/2) + 0.5  (exact identity, branch-free)
    return __builtin_fmaf(0.5f, fast_tanh(0.5f * x), 0.5f);
}
__device__ __forceinline__ float tern(float v) {
    // forward value of the straight-through ternary quantizer
    float s = 1.5f * fast_tanh(v) + 0.5f * fast_tanh(-3.f * v);
    return rintf(s);  // v_rndne_f32
}

// ---------------- fragment helpers (CDNA5 WMMA layouts, wave32) -------------
// A (f16, 16x32): lane: m = lane&15, kb = (lane>>4)*8 ; element e:
//   e in [0,8):  k = kb + e ; e in [8,16): k = kb + e + 8
// C/D (f32, 16x16): lane: n = lane&15 ; vgpr i: m = i + 8*(lane>>4)
// B (f16, 32x16): mirror of A with lane = column n.
__device__ __forceinline__ int kpat(int e, int kb) {
    return (e < 8) ? (kb + e) : (kb + e + 8);
}

__device__ __forceinline__ v16h a_frag_g(const float* __restrict__ base, int ld, int k0, int kmax) {
    int lane = threadIdx.x & 31;
    int m = lane & 15;
    int kb = (lane >> 4) * 8;
    v16h a;
#pragma unroll
    for (int e = 0; e < 16; ++e) {
        int k = k0 + kpat(e, kb);
        a[e] = (k < kmax) ? (_Float16)base[(size_t)m * ld + k] : (_Float16)0.f;
    }
    return a;
}

__device__ __forceinline__ v16h a_frag_lds(const float* tile, int ld, int Kv) {
    int lane = threadIdx.x & 31;
    int m = lane & 15;
    int kb = (lane >> 4) * 8;
    v16h a;
#pragma unroll
    for (int e = 0; e < 16; ++e) {
        int k = kpat(e, kb);
        a[e] = (k < Kv) ? (_Float16)tile[m * ld + k] : (_Float16)0.f;
    }
    return a;
}

__device__ __forceinline__ v16h b_frag(const float* __restrict__ W, int out_d, int in_d,
                                       int n0, int k0) {
    int lane = threadIdx.x & 31;
    int nn = n0 + (lane & 15);
    int kb = (lane >> 4) * 8;
    v16h b;
#pragma unroll
    for (int e = 0; e < 16; ++e) {
        int k = k0 + kpat(e, kb);
        float v = (nn < out_d && k < in_d) ? W[(size_t)nn * in_d + k] : 0.f;
        b[e] = (_Float16)v;
    }
    return b;
}

__device__ __forceinline__ v8f wmma16(v16h a, v16h b, v8f c) {
    return __builtin_amdgcn_wmma_f32_16x16x32_f16(false, a, false, b, (short)0, c, false, false);
}

// ---------------- Phase 1: observation QBN encoder --------------------------
__global__ void __launch_bounds__(32)
obs_encoder(const float* __restrict__ x,
            const float* __restrict__ w0, const float* __restrict__ b0,
            const float* __restrict__ w1, const float* __restrict__ b1,
            const float* __restrict__ w2, const float* __restrict__ b2,
            _Float16* __restrict__ xq) {
    __shared__ float tile[16 * 32];
    const int lane = threadIdx.x;
    const int n = lane & 15;
    const int half = lane >> 4;
    const size_t row0 = (size_t)blockIdx.x * 16;

    v16h B00 = b_frag(w0, OBS_MID, IN_DIM, 0, 0);
    v16h B01 = b_frag(w0, OBS_MID, IN_DIM, 0, 32);
    v16h B10 = b_frag(w0, OBS_MID, IN_DIM, 16, 0);
    v16h B11 = b_frag(w0, OBS_MID, IN_DIM, 16, 32);
    v16h B20 = b_frag(w1, OBS_MID, OBS_MID, 0, 0);
    v16h B21 = b_frag(w1, OBS_MID, OBS_MID, 16, 0);
    v16h B3  = b_frag(w2, OBS_BN, OBS_MID, 0, 0);

    v8f z8 = {};
    v16h A0 = a_frag_g(x + row0 * IN_DIM, IN_DIM, 0, IN_DIM);
    v16h A1 = a_frag_g(x + row0 * IN_DIM, IN_DIM, 32, IN_DIM);
    v8f c0 = wmma16(A1, B01, wmma16(A0, B00, z8));
    v8f c1 = wmma16(A1, B11, wmma16(A0, B10, z8));

    float bb0 = b0[n];
    float bb1 = (16 + n < OBS_MID) ? b0[16 + n] : 0.f;
#pragma unroll
    for (int i = 0; i < 8; ++i) {
        int m = i + 8 * half;
        tile[m * 32 + n] = fast_tanh(c0[i] + bb0);
        tile[m * 32 + 16 + n] = (16 + n < OBS_MID) ? fast_tanh(c1[i] + bb1) : 0.f;
    }
    __syncthreads();

    v16h Ah = a_frag_lds(tile, 32, OBS_MID);
    c0 = wmma16(Ah, B20, z8);
    c1 = wmma16(Ah, B21, z8);
    __syncthreads();
    float bb2 = b1[n];
    float bb3 = (16 + n < OBS_MID) ? b1[16 + n] : 0.f;
#pragma unroll
    for (int i = 0; i < 8; ++i) {
        int m = i + 8 * half;
        tile[m * 32 + n] = fast_tanh(c0[i] + bb2);
        tile[m * 32 + 16 + n] = (16 + n < OBS_MID) ? fast_tanh(c1[i] + bb3) : 0.f;
    }
    __syncthreads();

    Ah = a_frag_lds(tile, 32, OBS_MID);
    c0 = wmma16(Ah, B3, z8);
    if (n < OBS_BN) {
        float bb = b2[n];
#pragma unroll
        for (int i = 0; i < 8; ++i) {
            size_t row = row0 + i + 8 * half;
            xq[row * OBS_BN + n] = (_Float16)tern(c0[i] + bb);
        }
    }
}

// ---------------- Phase 2: recurrent GRU + QBN scan -------------------------
__global__ void __launch_bounds__(32)
gru_scan(const _Float16* __restrict__ xq,
         const float* __restrict__ wih, const float* __restrict__ whh,
         const float* __restrict__ bih, const float* __restrict__ bhh,
         const float* __restrict__ m0w, const float* __restrict__ m0b,
         const float* __restrict__ m1w, const float* __restrict__ m1b,
         const float* __restrict__ m2w, const float* __restrict__ m2b,
         const float* __restrict__ q0w, const float* __restrict__ q0b,
         const float* __restrict__ q1w, const float* __restrict__ q1b,
         const float* __restrict__ q2w, const float* __restrict__ q2b,
         const float* __restrict__ aw, const float* __restrict__ ab,
         float* __restrict__ out) {
    __shared__ float tile[16 * 16];
    const int lane = threadIdx.x;
    const int n = lane & 15;
    const int half = lane >> 4;
    const int bt = blockIdx.x;

    v16h Bih0 = b_frag(wih, 48, OBS_BN, 0, 0);
    v16h Bih1 = b_frag(wih, 48, OBS_BN, 16, 0);
    v16h Bih2 = b_frag(wih, 48, OBS_BN, 32, 0);
    v16h Bhh0 = b_frag(whh, 48, M_SIZE, 0, 0);
    v16h Bhh1 = b_frag(whh, 48, M_SIZE, 16, 0);
    v16h Bhh2 = b_frag(whh, 48, M_SIZE, 32, 0);
    v16h Bm0 = b_frag(m0w, 16, 16, 0, 0);
    v16h Bm1 = b_frag(m1w, 16, 16, 0, 0);
    v16h Bm2 = b_frag(m2w, 8, 16, 0, 0);
    v16h Bq0 = b_frag(q0w, 16, 8, 0, 0);
    v16h Bq1 = b_frag(q1w, 16, 16, 0, 0);
    v16h Bq2 = b_frag(q2w, 16, 16, 0, 0);
    v16h Ba  = b_frag(aw, 16, 16, 0, 0);

    float b_ir = bih[n], b_iz = bih[16 + n], b_in = bih[32 + n];
    float b_hr = bhh[n], b_hz = bhh[16 + n], b_hn = bhh[32 + n];
    float bm0 = m0b[n], bm1 = m1b[n], bm2 = (n < 8) ? m2b[n] : 0.f;
    float bq0 = q0b[n], bq1 = q1b[n], bq2 = q2b[n];
    float bact = ab[n];

    v8f hid = {};
    v8f z8 = {};

    for (int t = 0; t < TT; ++t) {
        // xq A-fragment: lanes 0..15 hold K=0..7 in elements 0..7; rest zero.
        v16h Ax = {};
        if (lane < 16) {
            const _Float16* p = xq + ((size_t)t * BB + (size_t)bt * 16 + lane) * OBS_BN;
#pragma unroll
            for (int e = 0; e < 8; ++e) Ax[e] = p[e];
        }
        v8f gir = wmma16(Ax, Bih0, z8);
        v8f giz = wmma16(Ax, Bih1, z8);
        v8f gin = wmma16(Ax, Bih2, z8);

        __syncthreads();
#pragma unroll
        for (int i = 0; i < 8; ++i) tile[(i + 8 * half) * 16 + n] = hid[i];
        __syncthreads();
        v16h Ah = a_frag_lds(tile, 16, M_SIZE);

        v8f ghr = wmma16(Ah, Bhh0, z8);
        v8f ghz = wmma16(Ah, Bhh1, z8);
        v8f ghn = wmma16(Ah, Bhh2, z8);

        v8f ht;
#pragma unroll
        for (int i = 0; i < 8; ++i) {
            float r = sigm(gir[i] + b_ir + ghr[i] + b_hr);
            float zz = sigm(giz[i] + b_iz + ghz[i] + b_hz);
            float nn = fast_tanh(gin[i] + b_in + r * (ghn[i] + b_hn));
            ht[i] = (1.f - zz) * nn + zz * hid[i];
        }

        // m2q: tanh, tanh, ternary
        __syncthreads();
#pragma unroll
        for (int i = 0; i < 8; ++i) tile[(i + 8 * half) * 16 + n] = ht[i];
        __syncthreads();
        v8f q = wmma16(a_frag_lds(tile, 16, M_SIZE), Bm0, z8);
#pragma unroll
        for (int i = 0; i < 8; ++i) q[i] = fast_tanh(q[i] + bm0);
        __syncthreads();
#pragma unroll
        for (int i = 0; i < 8; ++i) tile[(i + 8 * half) * 16 + n] = q[i];
        __syncthreads();
        q = wmma16(a_frag_lds(tile, 16, 16), Bm1, z8);
#pragma unroll
        for (int i = 0; i < 8; ++i) q[i] = fast_tanh(q[i] + bm1);
        __syncthreads();
#pragma unroll
        for (int i = 0; i < 8; ++i) tile[(i + 8 * half) * 16 + n] = q[i];
        __syncthreads();
        q = wmma16(a_frag_lds(tile, 16, 16), Bm2, z8);
#pragma unroll
        for (int i = 0; i < 8; ++i) q[i] = tern(q[i] + bm2);  // cols >= 8 masked by K=8 below

        // q2m: three tanh layers -> new hidden
        __syncthreads();
#pragma unroll
        for (int i = 0; i < 8; ++i) tile[(i + 8 * half) * 16 + n] = q[i];
        __syncthreads();
        v8f m = wmma16(a_frag_lds(tile, 16, OBS_BN), Bq0, z8);  // K=8 masks junk cols
#pragma unroll
        for (int i = 0; i < 8; ++i) m[i] = fast_tanh(m[i] + bq0);
        __syncthreads();
#pragma unroll
        for (int i = 0; i < 8; ++i) tile[(i + 8 * half) * 16 + n] = m[i];
        __syncthreads();
        m = wmma16(a_frag_lds(tile, 16, 16), Bq1, z8);
#pragma unroll
        for (int i = 0; i < 8; ++i) m[i] = fast_tanh(m[i] + bq1);
        __syncthreads();
#pragma unroll
        for (int i = 0; i < 8; ++i) tile[(i + 8 * half) * 16 + n] = m[i];
        __syncthreads();
        m = wmma16(a_frag_lds(tile, 16, 16), Bq2, z8);
#pragma unroll
        for (int i = 0; i < 8; ++i) m[i] = fast_tanh(m[i] + bq2);
        hid = m;

        // action head on new hidden, fused store
        __syncthreads();
#pragma unroll
        for (int i = 0; i < 8; ++i) tile[(i + 8 * half) * 16 + n] = m[i];
        __syncthreads();
        v8f y = wmma16(a_frag_lds(tile, 16, 16), Ba, z8);
#pragma unroll
        for (int i = 0; i < 8; ++i) {
            size_t b = (size_t)bt * 16 + i + 8 * half;
            out[((size_t)t * BB + b) * 16 + n] = fast_tanh(y[i] + bact);
        }
    }
}

// ---------------- host launch ----------------------------------------------
extern "C" void kernel_launch(void* const* d_in, const int* in_sizes, int n_in,
                              void* d_out, int out_size, void* d_ws, size_t ws_size,
                              hipStream_t stream) {
    const float* x      = (const float*)d_in[0];
    const float* obs_w0 = (const float*)d_in[1];
    const float* obs_b0 = (const float*)d_in[2];
    const float* obs_w1 = (const float*)d_in[3];
    const float* obs_b1 = (const float*)d_in[4];
    const float* obs_w2 = (const float*)d_in[5];
    const float* obs_b2 = (const float*)d_in[6];
    const float* gru_wih = (const float*)d_in[7];
    const float* gru_whh = (const float*)d_in[8];
    const float* gru_bih = (const float*)d_in[9];
    const float* gru_bhh = (const float*)d_in[10];
    const float* m2q_w0 = (const float*)d_in[11];
    const float* m2q_b0 = (const float*)d_in[12];
    const float* m2q_w1 = (const float*)d_in[13];
    const float* m2q_b1 = (const float*)d_in[14];
    const float* m2q_w2 = (const float*)d_in[15];
    const float* m2q_b2 = (const float*)d_in[16];
    const float* q2m_w0 = (const float*)d_in[17];
    const float* q2m_b0 = (const float*)d_in[18];
    const float* q2m_w1 = (const float*)d_in[19];
    const float* q2m_b1 = (const float*)d_in[20];
    const float* q2m_w2 = (const float*)d_in[21];
    const float* q2m_b2 = (const float*)d_in[22];
    const float* act_w  = (const float*)d_in[23];
    const float* act_b  = (const float*)d_in[24];

    _Float16* xq = (_Float16*)d_ws;  // [T*B, 8] f16 = 16 MB

    obs_encoder<<<(TT * BB) / 16, 32, 0, stream>>>(x, obs_w0, obs_b0, obs_w1, obs_b1,
                                                   obs_w2, obs_b2, xq);

    gru_scan<<<BB / 16, 32, 0, stream>>>(xq, gru_wih, gru_whh, gru_bih, gru_bhh,
                                         m2q_w0, m2q_b0, m2q_w1, m2q_b1, m2q_w2, m2q_b2,
                                         q2m_w0, q2m_b0, q2m_w1, q2m_b1, q2m_w2, q2m_b2,
                                         act_w, act_b, (float*)d_out);
}
